// SAGEModelBen_27152783245336
// MI455X (gfx1250) — matrix operand order repaired and checked by reading, not compiled, not run on hardware
//
#include <hip/hip_runtime.h>
#include <hip/hip_bf16.h>
#include <math.h>

typedef _Float16 h8  __attribute__((ext_vector_type(8)));
typedef _Float16 v16h __attribute__((ext_vector_type(16)));
typedef float    v8f  __attribute__((ext_vector_type(8)));

#define NODES 100000
#define EDGES 1600000
#define DIMF  128
#define NCLS  40

// ---------------- utility kernels ----------------

__global__ void zero_f32(float* __restrict__ p, int n) {
  int i = blockIdx.x * blockDim.x + threadIdx.x;
  if (i < n) p[i] = 0.0f;
}

__global__ void cvt_f16(_Float16* __restrict__ d, const float* __restrict__ s, int n) {
  int i = blockIdx.x * blockDim.x + threadIdx.x;
  if (i < n) d[i] = (_Float16)s[i];
}

// one wave (32 lanes) per edge; each lane moves 4 floats; lane0 counts degree.
__global__ void scatter_add(const float* __restrict__ x, const int* __restrict__ src,
                            const int* __restrict__ dst, float* __restrict__ sum,
                            float* __restrict__ cnt, int nE) {
  int t = blockIdx.x * blockDim.x + threadIdx.x;
  int e = t >> 5;
  int lane = t & 31;
  if (e >= nE) return;
  int s = src[e];
  int d = dst[e];
  const float4 v = *(const float4*)(x + (size_t)s * DIMF + lane * 4);
  float* o = sum + (size_t)d * DIMF + lane * 4;
  atomicAdd(o + 0, v.x);
  atomicAdd(o + 1, v.y);
  atomicAdd(o + 2, v.z);
  atomicAdd(o + 3, v.w);
  if (lane == 0) atomicAdd(cnt + d, 1.0f);
}

// aggh[i] = f16( sum[i] / max(cnt[row],1) )
__global__ void agg_prep(_Float16* __restrict__ aggh, const float* __restrict__ sum,
                         const float* __restrict__ cnt, int n) {
  int i = blockIdx.x * blockDim.x + threadIdx.x;
  if (i >= n) return;
  float c = cnt[i >> 7];
  float inv = (c > 1.0f) ? (1.0f / c) : 1.0f;
  aggh[i] = (_Float16)(sum[i] * inv);
}

// ---------------- fused dual-GEMM + bias + ReLU (WMMA) ----------------
// out[m,n] = relu( sum_k Aagg[m,k]*Wl[n,k] + bias[n] + sum_k Ax[m,k]*Wr[n,k] )
// Block: 256 thr = 8 waves. Block owns 16 rows x 128 cols; wave w owns cols [16w,16w+16).
// K = 128 -> 4 k-steps of 32, 2 WMMAs per step into one f32 accumulator.
__global__ void __launch_bounds__(256)
sage_gemm(const _Float16* __restrict__ Aagg, const _Float16* __restrict__ Ax,
          const _Float16* __restrict__ Wl,   const _Float16* __restrict__ Wr,
          const float* __restrict__ bias,
          float* __restrict__ outF, _Float16* __restrict__ outH, int nrows) {
  const int lane = threadIdx.x & 31;
  const int wave = threadIdx.x >> 5;          // 0..7 : which 16-col tile
  const int m0 = blockIdx.x * 16;
  const int n0 = wave * 16;

  // A fragment (16x32 f16): lane<16 holds M=lane, K={k..k+7} U {k+16..k+23};
  //                         lane>=16 holds M=lane-16, K={k+8..k+15} U {k+24..k+31}.
  int mrow = m0 + (lane & 15);
  if (mrow >= nrows) mrow = nrows - 1;        // clamped read (writes are guarded)
  const int ka = (lane >> 4) * 8;
  // B fragment (32x16 f16): lane<16 holds col N=lane, K={k..k+7} U {k+16..k+23};
  //                         lane>=16 holds col N=lane-16, K={k+8..k+15} U {k+24..k+31}.
  const int ncol = n0 + (lane & 15);
  const int kb = (lane >> 4) * 8;

  const _Float16* pAagg = Aagg + (size_t)mrow * DIMF + ka;
  const _Float16* pAx   = Ax   + (size_t)mrow * DIMF + ka;
  const _Float16* pWl   = Wl   + (size_t)ncol * DIMF + kb;
  const _Float16* pWr   = Wr   + (size_t)ncol * DIMF + kb;

  v8f acc = {};
#pragma unroll
  for (int ks = 0; ks < DIMF; ks += 32) {
    union { v16h v; h8 h[2]; } a0, a1, bl, br;
    a0.h[0] = *(const h8*)(pAagg + ks);
    a0.h[1] = *(const h8*)(pAagg + ks + 16);
    a1.h[0] = *(const h8*)(pAx + ks);
    a1.h[1] = *(const h8*)(pAx + ks + 16);
    bl.h[0] = *(const h8*)(pWl + ks);
    bl.h[1] = *(const h8*)(pWl + ks + 16);
    br.h[0] = *(const h8*)(pWr + ks);
    br.h[1] = *(const h8*)(pWr + ks + 16);
    acc = __builtin_amdgcn_wmma_f32_16x16x32_f16(false, a0.v, false, bl.v,
                                                 (short)0, acc, false, false);
    acc = __builtin_amdgcn_wmma_f32_16x16x32_f16(false, a1.v, false, br.v,
                                                 (short)0, acc, false, false);
  }

  // C/D layout: lane holds col (lane&15); VGPR r holds row m0 + r + 8*(lane>=16).
  const float bv = bias[ncol];
  const int rbase = m0 + ((lane >> 4) * 8);
#pragma unroll
  for (int r = 0; r < 8; ++r) {
    const int row = rbase + r;
    if (row < nrows) {
      float v = acc[r] + bv;
      v = v > 0.0f ? v : 0.0f;
      outF[(size_t)row * DIMF + ncol] = v;
      if (outH) outH[(size_t)row * DIMF + ncol] = (_Float16)v;
    }
  }
}

// ---------------- classifier + log_softmax: one wave per node ----------------
__global__ void __launch_bounds__(256)
classify(const float* __restrict__ h, const float* __restrict__ wc,
         const float* __restrict__ bc, float* __restrict__ out, int nrows) {
  const int lane = threadIdx.x & 31;
  const int wave = threadIdx.x >> 5;
  const int node = blockIdx.x * 8 + wave;
  if (node >= nrows) return;

  const float4 hv = *(const float4*)(h + (size_t)node * DIMF + lane * 4);
  float l0 = 0.0f;          // logit for class c == lane       (lane 0..31)
  float l1 = -1e30f;        // logit for class c == lane + 32  (lane 0..7)

  for (int c = 0; c < NCLS; ++c) {
    const float4 wv = *(const float4*)(wc + (size_t)c * DIMF + lane * 4);
    float p = hv.x * wv.x + hv.y * wv.y + hv.z * wv.z + hv.w * wv.w;
#pragma unroll
    for (int off = 16; off > 0; off >>= 1) p += __shfl_xor(p, off, 32);
    p += bc[c];
    if (c < 32) { if (lane == c)      l0 = p; }
    else        { if (lane == c - 32) l1 = p; }
  }

  float m = l0 > l1 ? l0 : l1;
#pragma unroll
  for (int off = 16; off > 0; off >>= 1) {
    float o = __shfl_xor(m, off, 32);
    m = m > o ? m : o;
  }
  float e = expf(l0 - m) + (lane < 8 ? expf(l1 - m) : 0.0f);
#pragma unroll
  for (int off = 16; off > 0; off >>= 1) e += __shfl_xor(e, off, 32);
  const float lse = logf(e) + m;

  out[(size_t)node * NCLS + lane] = l0 - lse;
  if (lane < 8) out[(size_t)node * NCLS + 32 + lane] = l1 - lse;
}

// ---------------- launch ----------------

extern "C" void kernel_launch(void* const* d_in, const int* in_sizes, int n_in,
                              void* d_out, int out_size, void* d_ws, size_t ws_size,
                              hipStream_t stream) {
  (void)in_sizes; (void)n_in; (void)out_size; (void)ws_size;
  const float* x    = (const float*)d_in[0];
  const int*   edge = (const int*)d_in[1];
  const float* w1_l = (const float*)d_in[2];
  const float* b1   = (const float*)d_in[3];
  const float* w1_r = (const float*)d_in[4];
  const float* w2_l = (const float*)d_in[5];
  const float* b2   = (const float*)d_in[6];
  const float* w2_r = (const float*)d_in[7];
  const float* wc   = (const float*)d_in[8];
  const float* bc   = (const float*)d_in[9];
  float* out = (float*)d_out;

  const int* src = edge;
  const int* dst = edge + EDGES;

  // workspace carve-out (256B-aligned chunks)
  char* ws = (char*)d_ws;
  size_t off = 0;
  auto take = [&](size_t bytes) -> void* {
    void* p = ws + off;
    off += (bytes + 255) & ~(size_t)255;
    return p;
  };
  _Float16* xh    = (_Float16*)take((size_t)NODES * DIMF * sizeof(_Float16));
  _Float16* aggh  = (_Float16*)take((size_t)NODES * DIMF * sizeof(_Float16));
  _Float16* w1l_h = (_Float16*)take((size_t)DIMF * DIMF * sizeof(_Float16));
  _Float16* w1r_h = (_Float16*)take((size_t)DIMF * DIMF * sizeof(_Float16));
  _Float16* w2l_h = (_Float16*)take((size_t)DIMF * DIMF * sizeof(_Float16));
  _Float16* w2r_h = (_Float16*)take((size_t)DIMF * DIMF * sizeof(_Float16));
  float*    sum   = (float*)take((size_t)NODES * DIMF * sizeof(float));   // 51.2MB (mult of 256)
  float*    cnt   = (float*)take((size_t)NODES * sizeof(float));          // contiguous after sum
  float*    h1f   = (float*)take((size_t)NODES * DIMF * sizeof(float));
  _Float16* h1h   = (_Float16*)take((size_t)NODES * DIMF * sizeof(_Float16));
  float*    h2f   = (float*)take((size_t)NODES * DIMF * sizeof(float));

  const int NF = NODES * DIMF;            // 12,800,000
  const int T = 256;
  const int gNF   = (NF + T - 1) / T;
  const int gZero = (NF + NODES + T - 1) / T;     // zeros sum + cnt in one shot
  const int gW    = (DIMF * DIMF + T - 1) / T;
  const int gScat = ((EDGES * 32) + T - 1) / T;
  const int gGemm = (NODES + 15) / 16;
  const int gCls  = (NODES + 7) / 8;

  // precision prep
  cvt_f16<<<gNF, T, 0, stream>>>(xh, x, NF);
  cvt_f16<<<gW, T, 0, stream>>>(w1l_h, w1_l, DIMF * DIMF);
  cvt_f16<<<gW, T, 0, stream>>>(w1r_h, w1_r, DIMF * DIMF);
  cvt_f16<<<gW, T, 0, stream>>>(w2l_h, w2_l, DIMF * DIMF);
  cvt_f16<<<gW, T, 0, stream>>>(w2r_h, w2_r, DIMF * DIMF);

  // ---- layer 1 ----
  zero_f32<<<gZero, T, 0, stream>>>(sum, NF + NODES);
  scatter_add<<<gScat, T, 0, stream>>>(x, src, dst, sum, cnt, EDGES);
  agg_prep<<<gNF, T, 0, stream>>>(aggh, sum, cnt, NF);
  sage_gemm<<<gGemm, T, 0, stream>>>(aggh, xh, w1l_h, w1r_h, b1, h1f, h1h, NODES);

  // ---- layer 2 ----
  zero_f32<<<gZero, T, 0, stream>>>(sum, NF + NODES);
  scatter_add<<<gScat, T, 0, stream>>>(h1f, src, dst, sum, cnt, EDGES);
  agg_prep<<<gNF, T, 0, stream>>>(aggh, sum, cnt, NF);
  sage_gemm<<<gGemm, T, 0, stream>>>(aggh, h1h, w2l_h, w2r_h, b2, h2f, (_Float16*)nullptr, NODES);

  // ---- classifier + log_softmax ----
  classify<<<gCls, T, 0, stream>>>(h2f, wc, bc, out, NODES);
}
